// SingleAttention_38560216384222
// MI455X (gfx1250) — matrix operand order repaired and checked by reading, not compiled, not run on hardware
//
#include <hip/hip_runtime.h>

// CDNA5 / gfx1250 WMMA types
typedef __attribute__((ext_vector_type(16))) _Float16 v16h;
typedef __attribute__((ext_vector_type(8)))  _Float16 v8h;
typedef __attribute__((ext_vector_type(4)))  _Float16 v4h;
typedef __attribute__((ext_vector_type(8)))  float    v8f;

#define BM 128
#define BN 128
#define BK 32
#define LDSP 40     // padded LDS row stride (halfs)
#define SSH 11      // log2(S), S = 2048
#define SMASK 2047

union HalfFrag { v16h v; v8h h[2]; };

// ---------------------------------------------------------------------------
// NT GEMM: C[b][m][n] = scale * ( sum_k A[m,k] * B[n,k] + bias[n] )
// fp32 in memory -> f16 in LDS (double buffered) -> f32 WMMA accumulate.
//  AMODE 1: A has (S+2) rows per batch; physical row = (m>>11)*2050 + (m&2047) + tap
//  BMODE 0: B row-major (n*ldb + k + tap*tapStrideB)   (repacked conv weights)
//  BMODE 2: B transposed (k*ldb + n), taps==1
//  CMODE 1: write into padded rows (m>>11)*2050 + (m&2047) + 2
//  MASK  1: strictly-causal: col >= row -> -1e9
// ---------------------------------------------------------------------------
template<int AMODE, int BMODE, int CMODE, int MASK>
__global__ __launch_bounds__(256) void gemm_f16wmma(
    const float* __restrict__ A, long sAb, int lda,
    const float* __restrict__ Bm, long sBb, int ldb, long tapStrideB,
    float* __restrict__ C, long sCb, int ldc,
    const float* __restrict__ bias,
    int kcShift, int taps, float scale)
{
  __shared__ alignas(16) _Float16 As[2][BM * LDSP];
  __shared__ alignas(16) _Float16 Bs[2][BN * LDSP];

  const int tid   = threadIdx.x;
  const int bx = blockIdx.x, by = blockIdx.y, bz = blockIdx.z;
  const int lane  = tid & 31;
  const int wid   = tid >> 5;   // 8 waves
  const int waveM = wid >> 2;   // 0..1  -> 64 rows
  const int waveN = wid & 3;    // 0..3  -> 32 cols
  const int hsel  = lane >> 4;
  const int rIn   = lane & 15;

  const float* Ab = A  + (long)bz * sAb;
  const float* Bb = Bm + (long)bz * sBb;
  float*       Cb = C  + (long)bz * sCb;

  // staging coordinates (thread-invariant)
  const int ar0 = tid >> 3;          // row base (+ i*32)
  const int ac  = (tid & 7) * 4;     // col (quad)
  const int bk0 = tid >> 5;          // BMODE2: k base (+ i*8)
  const int bn0 = (tid & 31) * 4;    // BMODE2: n quad

  float4 aR[4], bR[4];

  auto loadTiles = [&](int kt) {
    const int k0   = kt * BK;
    const int tap  = k0 >> kcShift;
    const int kin0 = k0 - (tap << kcShift);
    #pragma unroll
    for (int i = 0; i < 4; ++i) {
      const int r = ar0 + i * 32;
      long srow;
      if (AMODE == 1) { int grow = by * BM + r;
                        srow = (long)(grow >> SSH) * 2050 + (grow & SMASK) + tap; }
      else            { srow = by * BM + r; }
      aR[i] = *(const float4*)(Ab + srow * (long)lda + (kin0 + ac));
    }
    #pragma unroll
    for (int i = 0; i < 4; ++i) {
      if (BMODE == 2) {
        const int kk = bk0 + i * 8;
        bR[i] = *(const float4*)(Bb + (long)(kin0 + kk) * ldb + (bx * BN + bn0));
      } else {
        const int n = bx * BN + ar0 + i * 32;
        bR[i] = *(const float4*)(Bb + (long)n * ldb + (long)tap * tapStrideB + (kin0 + ac));
      }
    }
  };

  auto storeTiles = [&](int buf) {
    _Float16* Ad = As[buf];
    _Float16* Bd = Bs[buf];
    #pragma unroll
    for (int i = 0; i < 4; ++i) {
      v4h h; h[0]=(_Float16)aR[i].x; h[1]=(_Float16)aR[i].y; h[2]=(_Float16)aR[i].z; h[3]=(_Float16)aR[i].w;
      *(v4h*)&Ad[(ar0 + i * 32) * LDSP + ac] = h;
    }
    #pragma unroll
    for (int i = 0; i < 4; ++i) {
      if (BMODE == 2) {
        const int kk = bk0 + i * 8;
        Bd[(bn0 + 0) * LDSP + kk] = (_Float16)bR[i].x;
        Bd[(bn0 + 1) * LDSP + kk] = (_Float16)bR[i].y;
        Bd[(bn0 + 2) * LDSP + kk] = (_Float16)bR[i].z;
        Bd[(bn0 + 3) * LDSP + kk] = (_Float16)bR[i].w;
      } else {
        v4h h; h[0]=(_Float16)bR[i].x; h[1]=(_Float16)bR[i].y; h[2]=(_Float16)bR[i].z; h[3]=(_Float16)bR[i].w;
        *(v4h*)&Bd[(ar0 + i * 32) * LDSP + ac] = h;
      }
    }
  };

  v8f acc[4][2];
  for (int i = 0; i < 4; ++i)
    for (int j = 0; j < 2; ++j)
      for (int e = 0; e < 8; ++e) acc[i][j][e] = 0.0f;

  const int nSteps = (taps << kcShift) / BK;
  loadTiles(0);
  storeTiles(0);

  for (int kt = 0; kt < nSteps; ++kt) {
    __syncthreads();                       // LDS buf[cur] complete for all waves
    const int  cur  = kt & 1;
    const bool more = (kt + 1 < nSteps);
    if (more) loadTiles(kt + 1);           // prefetch next tile (hidden under WMMA)

    const _Float16* Ac = As[cur];
    const _Float16* Bc = Bs[cur];
    HalfFrag aF[4], bF[2];
    const int ac0 = hsel * 8;              // A: K = {c0..c0+7} U {c0+16..c0+23}
    #pragma unroll
    for (int mt = 0; mt < 4; ++mt) {
      int row = waveM * 64 + mt * 16 + rIn;
      aF[mt].h[0] = *(const v8h*)&Ac[row * LDSP + ac0];
      aF[mt].h[1] = *(const v8h*)&Ac[row * LDSP + ac0 + 16];
    }
    #pragma unroll
    for (int nt = 0; nt < 2; ++nt) {       // B: K = {c0..c0+7} U {c0+16..c0+23}, same as A
      int row = waveN * 32 + nt * 16 + rIn;
      bF[nt].h[0] = *(const v8h*)&Bc[row * LDSP + ac0];
      bF[nt].h[1] = *(const v8h*)&Bc[row * LDSP + ac0 + 16];
    }
    #pragma unroll
    for (int mt = 0; mt < 4; ++mt)
      #pragma unroll
      for (int nt = 0; nt < 2; ++nt)
        acc[mt][nt] = __builtin_amdgcn_wmma_f32_16x16x32_f16(
            false, aF[mt].v, false, bF[nt].v, (short)0, acc[mt][nt], false, false);

    if (more) storeTiles((kt + 1) & 1);    // safe: other buffer, fenced by next barrier
  }

  // epilogue: lane<16 -> M = vgpr, lane>=16 -> M = 8+vgpr; N = lane&15
  #pragma unroll
  for (int mt = 0; mt < 4; ++mt) {
    #pragma unroll
    for (int nt = 0; nt < 2; ++nt) {
      const int col = bx * BN + waveN * 32 + nt * 16 + rIn;
      const float bv = bias ? bias[col] : 0.0f;
      #pragma unroll
      for (int e = 0; e < 8; ++e) {
        const int row = by * BM + waveM * 64 + mt * 16 + hsel * 8 + e;
        float v = (acc[mt][nt][e] + bv) * scale;
        if (MASK && col >= row) v = -1e9f;
        long orow;
        if (CMODE == 1) orow = (long)(row >> SSH) * 2050 + (row & SMASK) + 2;
        else            orow = row;
        Cb[orow * (long)ldc + col] = v;
      }
    }
  }
}

// repack conv3 weights (Cout,Cin,3) -> (3,Cout,Cin) for contiguous K reads
__global__ void repack3(const float* __restrict__ w, float* __restrict__ o,
                        int Cin, int Cout)
{
  int k   = blockIdx.x * blockDim.x + threadIdx.x;
  int n   = blockIdx.y;
  int tap = blockIdx.z;
  o[((long)tap * Cout + n) * Cin + k] = w[((long)n * Cin + k) * 3 + tap];
}

// fill pad rows 0,1 of each (S+2)-row batch block with the bias vector
__global__ void padfill_rows(float* __restrict__ buf, const float* __restrict__ bias,
                             int Cw, int nb)
{
  int i = blockIdx.x * blockDim.x + threadIdx.x;
  int total = nb * 2 * Cw;
  if (i >= total) return;
  int b = i / (2 * Cw);
  int rem = i - b * (2 * Cw);
  int r = rem / Cw;
  int c = rem - r * Cw;
  buf[((long)b * 2050 + r) * (long)Cw + c] = bias[c];
}

// out = a * sigmoid(b) [+ res]   (vectorized float4; N = 1<<nShift)
__global__ void glu_combine(const float* __restrict__ a, int ldaa,
                            const float* __restrict__ g, int ldg,
                            const float* __restrict__ res, int ldr,
                            float* __restrict__ o, int ldo,
                            long total4, int nShift)
{
  long i = blockIdx.x * (long)blockDim.x + threadIdx.x;
  if (i >= total4) return;
  long e = i << 2;
  long r = e >> nShift;
  int  c = (int)(e & ((1 << nShift) - 1));
  float4 av = *(const float4*)(a + r * (long)ldaa + c);
  float4 gv = *(const float4*)(g + r * (long)ldg  + c);
  float4 v;
  v.x = av.x * (1.0f / (1.0f + __expf(-gv.x)));
  v.y = av.y * (1.0f / (1.0f + __expf(-gv.y)));
  v.z = av.z * (1.0f / (1.0f + __expf(-gv.z)));
  v.w = av.w * (1.0f / (1.0f + __expf(-gv.w)));
  if (res) {
    float4 rv = *(const float4*)(res + r * (long)ldr + c);
    v.x += rv.x; v.y += rv.y; v.z += rv.z; v.w += rv.w;
  }
  *(float4*)(o + r * (long)ldo + c) = v;
}

// in-place row softmax over S columns (one block per row)
__global__ __launch_bounds__(256) void softmax_rows(float* __restrict__ w, int S)
{
  __shared__ float red[256];
  float4* p = (float4*)(w + (long)blockIdx.x * S);
  const int n4 = S >> 2;
  const int tid = threadIdx.x;
  float mx = -3.4e38f;
  for (int i = tid; i < n4; i += 256) {
    float4 v = p[i];
    mx = fmaxf(mx, fmaxf(fmaxf(v.x, v.y), fmaxf(v.z, v.w)));
  }
  red[tid] = mx; __syncthreads();
  for (int s = 128; s > 0; s >>= 1) { if (tid < s) red[tid] = fmaxf(red[tid], red[tid + s]); __syncthreads(); }
  mx = red[0]; __syncthreads();
  float sum = 0.f;
  for (int i = tid; i < n4; i += 256) {
    float4 v = p[i];
    v.x = __expf(v.x - mx); v.y = __expf(v.y - mx);
    v.z = __expf(v.z - mx); v.w = __expf(v.w - mx);
    sum += v.x + v.y + v.z + v.w;
    p[i] = v;
  }
  red[tid] = sum; __syncthreads();
  for (int s = 128; s > 0; s >>= 1) { if (tid < s) red[tid] += red[tid + s]; __syncthreads(); }
  float inv = 1.0f / red[0];
  for (int i = tid; i < n4; i += 256) {
    float4 v = p[i];
    v.x *= inv; v.y *= inv; v.z *= inv; v.w *= inv;
    p[i] = v;
  }
}

// ---------------------------------------------------------------------------
static inline void gemm(hipStream_t st, int M, int N, int nb,
    const float* A, long sAb, int lda, int aMode,
    const float* B, long sBb, int ldb, long tapStrideB, int bMode,
    float* C, long sCb, int ldc, int cMode,
    const float* bias, int Kc, int taps, float scale, int mask)
{
  dim3 grid(N / BN, M / BM, nb);
  int kcShift = 31 - __builtin_clz((unsigned)Kc);
  if (aMode == 1)
    gemm_f16wmma<1,0,0,0><<<grid,256,0,st>>>(A,sAb,lda,B,sBb,ldb,tapStrideB,C,sCb,ldc,bias,kcShift,taps,scale);
  else if (bMode == 2)
    gemm_f16wmma<0,2,0,0><<<grid,256,0,st>>>(A,sAb,lda,B,sBb,ldb,tapStrideB,C,sCb,ldc,bias,kcShift,taps,scale);
  else if (cMode == 1)
    gemm_f16wmma<0,0,1,0><<<grid,256,0,st>>>(A,sAb,lda,B,sBb,ldb,tapStrideB,C,sCb,ldc,bias,kcShift,taps,scale);
  else if (mask)
    gemm_f16wmma<0,0,0,1><<<grid,256,0,st>>>(A,sAb,lda,B,sBb,ldb,tapStrideB,C,sCb,ldc,bias,kcShift,taps,scale);
  else
    gemm_f16wmma<0,0,0,0><<<grid,256,0,st>>>(A,sAb,lda,B,sBb,ldb,tapStrideB,C,sCb,ldc,bias,kcShift,taps,scale);
}

static inline void glu(hipStream_t st, long M, int N,
                       const float* a, int ldaa, const float* g, int ldg,
                       const float* res, int ldr, float* o, int ldo)
{
  long total4 = (M * (long)N) >> 2;
  int blocks = (int)((total4 + 255) / 256);
  int nShift = 31 - __builtin_clz((unsigned)N);
  glu_combine<<<blocks, 256, 0, st>>>(a, ldaa, g, ldg, res, ldr, o, ldo, total4, nShift);
}

extern "C" void kernel_launch(void* const* d_in, const int* in_sizes, int n_in,
                              void* d_out, int out_size, void* d_ws, size_t ws_size,
                              hipStream_t stream)
{
  (void)in_sizes; (void)n_in; (void)out_size; (void)ws_size;
  const int Bb = 4, S = 2048, D = 1024, OUTC = 1024;

  const float* X   = (const float*)d_in[0];
  const float* qw1 = (const float*)d_in[1];  const float* qb1 = (const float*)d_in[2];
  const float* qw2 = (const float*)d_in[3];  const float* qb2 = (const float*)d_in[4];
  const float* qw3 = (const float*)d_in[5];  const float* qb3 = (const float*)d_in[6];
  const float* outw = (const float*)d_in[51]; const float* outb = (const float*)d_in[52];

  float* ws = (float*)d_ws;
  const size_t PBUF = (size_t)Bb * S * OUTC;        // 8,388,608 floats
  const size_t SLOT = (size_t)Bb * (S + 2) * 2048;  // 16,793,600 floats
  float* qbuf = ws;
  float* kbuf = ws + PBUF;
  float* vbuf = ws + 2 * PBUF;
  float* abuf = ws + 3 * PBUF;
  float* T1 = ws + 4 * PBUF;
  float* T2 = T1 + SLOT;
  float* T3 = T2 + SLOT;
  float* T4 = T3 + SLOT;
  float* RP = T4 + SLOT;      // repacked conv3 weights (up to 3*2048*2048 floats)

  // ---------------- q = GLU-MLP(X) * 1/sqrt(64) ----------------
  gemm(stream, Bb*S, 4096, 1, X,0,D,0,     qw1,0,D,0,0,    T1,0,4096,0, qb1, D,   1, 1.f, 0);
  glu (stream, (long)Bb*S, 2048, T1,4096, T1+2048,4096, nullptr,0, T3,2048);
  gemm(stream, Bb*S, 2048, 1, T3,0,2048,0, qw2,0,2048,0,0, T4,0,2048,0, qb2, 2048,1, 1.f, 0);
  glu (stream, (long)Bb*S, 1024, T4,2048, T4+1024,2048, nullptr,0, T1,1024);
  gemm(stream, Bb*S, 1024, 1, T1,0,1024,0, qw3,0,1024,0,0, qbuf,0,1024,0, qb3, 1024,1, 0.125f, 0);

  // ---------------- k = glu_conv(X,kp), v = glu_conv(X,vp) ----------------
  for (int set = 0; set < 2; ++set) {
    const int base = 7 + set * 22;
    const float* Wp[22];
    for (int j = 0; j < 22; ++j) Wp[j] = (const float*)d_in[base + j];
    float* dst = (set == 0) ? kbuf : vbuf;

    // res1 = X @ res1_w^T + b                              -> T4 (B,S,2048)
    gemm(stream, Bb*S, 2048, 1, X,0,D,0, Wp[0],0,D,0,0, T4,0,2048,0, Wp[1], D,1, 1.f,0);
    // h1a = 1x1(c1a), staged with 2 bias pad-rows/batch    -> T1 (B,S+2,2048)
    gemm(stream, Bb*S, 2048, 1, X,0,D,0, Wp[4],0,D,0,0, T1,0,2048,1, Wp[5], D,1, 1.f,0);
    padfill_rows<<<(Bb*2*2048 + 255)/256, 256, 0, stream>>>(T1, Wp[5], 2048, Bb);
    // XaPre = conv3(c2a) over h1a (3-tap GEMM, K=3*2048)   -> T2
    repack3<<<dim3(2048/256, 2048, 3), 256, 0, stream>>>(Wp[8], RP, 2048, 2048);
    gemm(stream, Bb*S, 2048, 1, T1,0,2048,1, RP,0,2048,(long)2048*2048,0, T2,0,2048,0, Wp[9], 2048,3, 1.f,0);
    // h1b / XbPre
    gemm(stream, Bb*S, 2048, 1, X,0,D,0, Wp[6],0,D,0,0, T1,0,2048,1, Wp[7], D,1, 1.f,0);
    padfill_rows<<<(Bb*2*2048 + 255)/256, 256, 0, stream>>>(T1, Wp[7], 2048, Bb);
    repack3<<<dim3(2048/256, 2048, 3), 256, 0, stream>>>(Wp[10], RP, 2048, 2048);
    gemm(stream, Bb*S, 2048, 1, T1,0,2048,1, RP,0,2048,(long)2048*2048,0, T3,0,2048,0, Wp[11], 2048,3, 1.f,0);
    // X2 = Xa*sigmoid(Xb) + res1                           -> T1 (B,S,2048)
    glu(stream, (long)Bb*S, 2048, T2,2048, T3,2048, T4,2048, T1,2048);

    // res2 = X @ res2_w^T + b                              -> T4 (B,S,1024)
    gemm(stream, Bb*S, 1024, 1, X,0,D,0, Wp[2],0,D,0,0, T4,0,1024,0, Wp[3], D,1, 1.f,0);
    // h2a = 1x1(c3a) over X2, padded staging               -> T2 (B,S+2,1024)
    gemm(stream, Bb*S, 1024, 1, T1,0,2048,0, Wp[12],0,2048,0,0, T2,0,1024,1, Wp[13], 2048,1, 1.f,0);
    padfill_rows<<<(Bb*2*1024 + 255)/256, 256, 0, stream>>>(T2, Wp[13], 1024, Bb);
    // YaPre = conv3(c4a)                                   -> T3
    repack3<<<dim3(1024/256, 1024, 3), 256, 0, stream>>>(Wp[16], RP, 1024, 1024);
    gemm(stream, Bb*S, 1024, 1, T2,0,1024,1, RP,0,1024,(long)1024*1024,0, T3,0,1024,0, Wp[17], 1024,3, 1.f,0);
    // h2b / YbPre  (YbPre into second half of T2 slot)
    gemm(stream, Bb*S, 1024, 1, T1,0,2048,0, Wp[14],0,2048,0,0, T2,0,1024,1, Wp[15], 2048,1, 1.f,0);
    padfill_rows<<<(Bb*2*1024 + 255)/256, 256, 0, stream>>>(T2, Wp[15], 1024, Bb);
    float* T2h = T2 + (size_t)Bb * (S + 2) * 1024;
    repack3<<<dim3(1024/256, 1024, 3), 256, 0, stream>>>(Wp[18], RP, 1024, 1024);
    gemm(stream, Bb*S, 1024, 1, T2,0,1024,1, RP,0,1024,(long)1024*1024,0, T2h,0,1024,0, Wp[19], 1024,3, 1.f,0);
    // Y = Ya*sigmoid(Yb) + res2                            -> T1 (B,S,1024)
    glu(stream, (long)Bb*S, 1024, T3,1024, T2h,1024, T4,1024, T1,1024);
    // dst = Y @ lin_w^T + lin_b
    gemm(stream, Bb*S, 1024, 1, T1,0,1024,0, Wp[20],0,1024,0,0, dst,0,1024,0, Wp[21], 1024,1, 1.f,0);
  }

  // ---------------- attention ----------------
  float* Wout = (float*)d_out;          // attn (B,S,OUT)
  float* Pw   = (float*)d_out + PBUF;   // attn_weights (B,S,S), fp32 output

  // masked scores = q @ k^T  (strict causal), per batch
  gemm(stream, S, S, Bb, qbuf,(long)S*OUTC,OUTC,0, kbuf,(long)S*OUTC,OUTC,0,0,
       Pw,(long)S*S,S,0, nullptr, OUTC,1, 1.f, 1);
  // softmax rows in place (row 0 -> uniform 1/S, matching reference)
  softmax_rows<<<Bb*S, 256, 0, stream>>>(Pw, S);
  // attn = P @ v   (B transposed: v stored (kpos, d))
  gemm(stream, S, OUTC, Bb, Pw,(long)S*S,S,0, vbuf,(long)S*OUTC,OUTC,0,2,
       abuf,(long)S*OUTC,OUTC,0, nullptr, S,1, 1.f, 0);
  // out = attn @ out_w^T + out_b
  gemm(stream, Bb*S, OUTC, 1, abuf,0,OUTC,0, outw,0,OUTC,0,0, Wout,0,OUTC,0, outb, OUTC,1, 1.f, 0);
}
